// CrossAttention_77068893160018
// MI455X (gfx1250) — matrix operand order repaired and checked by reading, not compiled, not run on hardware
//
#include <hip/hip_runtime.h>

#define B_SZ    4
#define T_LEN   2048
#define S_LEN   2048
#define C_DIM   512
#define N_HEADS 16
#define D_HEAD  32
#define SCALE   0.17677669529663687f  // 1/sqrt(32)

typedef __attribute__((ext_vector_type(16))) __bf16 v16bf;
typedef __attribute__((ext_vector_type(8)))  __bf16 v8bf;
typedef __attribute__((ext_vector_type(8)))  float  v8f;

__device__ __forceinline__ v8f wmma_bf16(v16bf a, v16bf b, v8f c) {
  // (neg_a, A, neg_b, B, c_mod, C, reuse_a, reuse_b)
  return __builtin_amdgcn_wmma_f32_16x16x32_bf16(false, a, false, b, (short)0, c,
                                                 false, false);
}

__device__ __forceinline__ v16bf cat8(v8bf lo, v8bf hi) {
  v16bf r;
#pragma unroll
  for (int i = 0; i < 8; ++i) { r[i] = lo[i]; r[i + 8] = hi[i]; }
  return r;
}

__device__ __forceinline__ v16bf cvt16(v8f lo, v8f hi) {
  v16bf r;
#pragma unroll
  for (int i = 0; i < 8; ++i) { r[i] = (__bf16)lo[i]; r[i + 8] = (__bf16)hi[i]; }
  return r;
}

// ---------------------------------------------------------------------------
// Weight prep: Wt[mat][n][k] = (bf16) W_mat[k][n]   (LDS-tiled transpose)
// grid (16,16,4), block (32,8)
// ---------------------------------------------------------------------------
__global__ void k_prep(const float* __restrict__ Wq, const float* __restrict__ Wk,
                       const float* __restrict__ Wv, const float* __restrict__ Wo,
                       __bf16* __restrict__ Wt) {
  __shared__ float tile[32][33];
  const float* Ws[4] = {Wq, Wk, Wv, Wo};
  const float* W = Ws[blockIdx.z];
  const int tx = threadIdx.x, ty = threadIdx.y;
  const int bx = blockIdx.x, by = blockIdx.y;
#pragma unroll
  for (int i = 0; i < 4; ++i) {
    int r = ty + i * 8;
    tile[r][tx] = W[(size_t)(by * 32 + r) * C_DIM + bx * 32 + tx];
  }
  __syncthreads();
#pragma unroll
  for (int i = 0; i < 4; ++i) {
    int r = ty + i * 8;
    Wt[((size_t)blockIdx.z * C_DIM + bx * 32 + r) * C_DIM + by * 32 + tx] =
        (__bf16)tile[tx][r];
  }
}

// ---------------------------------------------------------------------------
// Fused Q/K/V projections. One wave computes a 16x64 output tile.
// waves = 3 mats * 512 rowblocks * 8 coltiles = 12288 -> 3072 blocks of 128.
// Q stored [B,H,T,32] (pre-scaled), K stored [B,H,S,32], V stored transposed
// [B,H,32,S] so the PV WMMA B-operand is a contiguous load.
// ---------------------------------------------------------------------------
__global__ void __launch_bounds__(128)
k_qkv(const float* __restrict__ x, const float* __restrict__ ctx,
      const __bf16* __restrict__ Wt, const float* __restrict__ bq,
      const float* __restrict__ bk, const float* __restrict__ bv,
      __bf16* __restrict__ Qs, __bf16* __restrict__ Ks, __bf16* __restrict__ Vt) {
  const int lane = threadIdx.x & 31;
  const int li = lane & 15, hi = lane >> 4;
  const int wave = blockIdx.x * 4 + (threadIdx.x >> 5);
  const int mat = wave / 4096;            // 0=Q,1=K,2=V
  const int rem = wave - mat * 4096;
  const int rowblk = rem >> 3;            // 0..511 (16-row blocks of B*T)
  const int ct = rem & 7;                 // 64-wide column tile

  const float* src  = (mat == 0) ? x : ctx;
  const float* bias = (mat == 0) ? bq : ((mat == 1) ? bk : bv);
  const float* ap = src + (size_t)(rowblk * 16 + li) * C_DIM;   // A row
  const __bf16* wb = Wt + (size_t)mat * C_DIM * C_DIM;

  v8f acc[4] = {{}, {}, {}, {}};
  for (int k0 = 0; k0 < C_DIM; k0 += 32) {
    v8f alo = *(const v8f*)(ap + k0 + 16 * hi);
    v8f ahi = *(const v8f*)(ap + k0 + 16 * hi + 8);
    v16bf a = cvt16(alo, ahi);
#pragma unroll
    for (int st = 0; st < 4; ++st) {
      int n = ct * 64 + st * 16 + li;
      v16bf bm = *(const v16bf*)(wb + (size_t)n * C_DIM + k0 + 16 * hi);
      acc[st] = wmma_bf16(a, bm, acc[st]);
    }
  }

#pragma unroll
  for (int st = 0; st < 4; ++st) {
#pragma unroll
    for (int r = 0; r < 8; ++r) {
      int m = rowblk * 16 + r + 8 * hi;   // global row in [0,8192)
      int t = m & 2047, bb = m >> 11;
      int n = ct * 64 + st * 16 + li;
      float v = acc[st][r] + bias[n];
      int hh = n >> 5, d = n & 31;
      if (mat == 0) {
        Qs[(((size_t)(bb * N_HEADS + hh) * T_LEN) + t) * D_HEAD + d] =
            (__bf16)(v * SCALE);
      } else if (mat == 1) {
        Ks[(((size_t)(bb * N_HEADS + hh) * S_LEN) + t) * D_HEAD + d] = (__bf16)v;
      } else {
        Vt[(((size_t)(bb * N_HEADS + hh) * D_HEAD) + d) * S_LEN + t] = (__bf16)v;
      }
    }
  }
}

// ---------------------------------------------------------------------------
// Flash attention: one wave per (b, h, 16-query block). Online softmax over
// S=2048 keys in chunks of 32 (2 QK WMMAs + 2 PV WMMAs per chunk).
// waves = 4*16*128 = 8192 -> 2048 blocks of 128.
// ---------------------------------------------------------------------------
__global__ void __launch_bounds__(128)
k_attn(const __bf16* __restrict__ Qs, const __bf16* __restrict__ Ks,
       const __bf16* __restrict__ Vt, __bf16* __restrict__ AO) {
  __shared__ __attribute__((aligned(64))) __bf16 Plds[4][16 * 32];
  const int wib = threadIdx.x >> 5;
  const int lane = threadIdx.x & 31;
  const int li = lane & 15, hi = lane >> 4;
  const int wave = blockIdx.x * 4 + wib;
  const int qb = wave & 127;
  const int h = (wave >> 7) & 15;
  const int b = wave >> 11;

  const __bf16* qrow =
      Qs + (((size_t)(b * N_HEADS + h) * T_LEN) + qb * 16 + li) * D_HEAD;
  const v16bf qa = cat8(*(const v8bf*)(qrow + 16 * hi),
                        *(const v8bf*)(qrow + 16 * hi + 8));

  const __bf16* kbase = Ks + (size_t)(b * N_HEADS + h) * S_LEN * D_HEAD;
  const __bf16* vbase = Vt + (size_t)(b * N_HEADS + h) * D_HEAD * S_LEN;
  __bf16* pw = &Plds[wib][0];

  float mrow[8], lrow[8];
#pragma unroll
  for (int r = 0; r < 8; ++r) { mrow[r] = -1e30f; lrow[r] = 0.0f; }
  v8f o0 = {}, o1 = {};

  for (int c = 0; c < S_LEN; c += 32) {
    v16bf kb0 = *(const v16bf*)(kbase + (size_t)(c + li) * D_HEAD + 16 * hi);
    v16bf kb1 = *(const v16bf*)(kbase + (size_t)(c + 16 + li) * D_HEAD + 16 * hi);
    v8f z = {};
    v8f s0 = wmma_bf16(qa, kb0, z);
    v8f s1 = wmma_bf16(qa, kb1, z);

    if (c + 32 < S_LEN) {
      __builtin_prefetch(kbase + (size_t)(c + 32 + li) * D_HEAD, 0, 3);
      __builtin_prefetch(vbase + (size_t)li * S_LEN + c + 32, 0, 3);
    }

    // Online softmax: row reductions across the 16 lanes of each half-wave.
#pragma unroll
    for (int r = 0; r < 8; ++r) {
      float mx = fmaxf(s0[r], s1[r]);
      mx = fmaxf(mx, __shfl_xor(mx, 1, 32));
      mx = fmaxf(mx, __shfl_xor(mx, 2, 32));
      mx = fmaxf(mx, __shfl_xor(mx, 4, 32));
      mx = fmaxf(mx, __shfl_xor(mx, 8, 32));
      float nm = fmaxf(mrow[r], mx);
      float al = __expf(mrow[r] - nm);
      mrow[r] = nm;
      float e0 = __expf(s0[r] - nm);
      float e1 = __expf(s1[r] - nm);
      s0[r] = e0; s1[r] = e1;
      float rs = e0 + e1;
      rs += __shfl_xor(rs, 1, 32);
      rs += __shfl_xor(rs, 2, 32);
      rs += __shfl_xor(rs, 4, 32);
      rs += __shfl_xor(rs, 8, 32);
      lrow[r] = lrow[r] * al + rs;
      o0[r] *= al;
      o1[r] *= al;
    }

    // C/D layout -> A layout via per-wave LDS round trip.
#pragma unroll
    for (int r = 0; r < 8; ++r) {
      pw[(r + 8 * hi) * 32 + li]      = (__bf16)s0[r];
      pw[(r + 8 * hi) * 32 + 16 + li] = (__bf16)s1[r];
    }
    asm volatile("s_wait_dscnt 0" ::: "memory");
    const __bf16* pr = pw + li * 32;
    v16bf pa = cat8(*(const v8bf*)(pr + 16 * hi), *(const v8bf*)(pr + 16 * hi + 8));

    v16bf vb0 = *(const v16bf*)(vbase + (size_t)li * S_LEN + c + 16 * hi);
    v16bf vb1 = *(const v16bf*)(vbase + (size_t)(16 + li) * S_LEN + c + 16 * hi);
    o0 = wmma_bf16(pa, vb0, o0);
    o1 = wmma_bf16(pa, vb1, o1);
  }

#pragma unroll
  for (int r = 0; r < 8; ++r) {
    float inv = 1.0f / lrow[r];
    int t = qb * 16 + r + 8 * hi;
    __bf16* dst = AO + (size_t)(b * T_LEN + t) * C_DIM + h * D_HEAD;
    dst[li]      = (__bf16)(o0[r] * inv);
    dst[16 + li] = (__bf16)(o1[r] * inv);
  }
}

// ---------------------------------------------------------------------------
// Output projection: out = AO(bf16) @ Wo + bo  (fp32 out).
// waves = 512 * 8 = 4096 -> 1024 blocks of 128.
// ---------------------------------------------------------------------------
__global__ void __launch_bounds__(128)
k_oproj(const __bf16* __restrict__ AO, const __bf16* __restrict__ Wt,
        const float* __restrict__ bo, float* __restrict__ out) {
  const int lane = threadIdx.x & 31;
  const int li = lane & 15, hi = lane >> 4;
  const int wave = blockIdx.x * 4 + (threadIdx.x >> 5);
  const int rowblk = wave >> 3;
  const int ct = wave & 7;
  const __bf16* ap = AO + (size_t)(rowblk * 16 + li) * C_DIM;
  const __bf16* wb = Wt + (size_t)3 * C_DIM * C_DIM;

  v8f acc[4] = {{}, {}, {}, {}};
  for (int k0 = 0; k0 < C_DIM; k0 += 32) {
    v16bf a = cat8(*(const v8bf*)(ap + k0 + 16 * hi),
                   *(const v8bf*)(ap + k0 + 16 * hi + 8));
#pragma unroll
    for (int st = 0; st < 4; ++st) {
      int n = ct * 64 + st * 16 + li;
      v16bf bm = *(const v16bf*)(wb + (size_t)n * C_DIM + k0 + 16 * hi);
      acc[st] = wmma_bf16(a, bm, acc[st]);
    }
  }
#pragma unroll
  for (int st = 0; st < 4; ++st) {
#pragma unroll
    for (int r = 0; r < 8; ++r) {
      int m = rowblk * 16 + r + 8 * hi;
      int n = ct * 64 + st * 16 + li;
      out[(size_t)m * C_DIM + n] = acc[st][r] + bo[n];
    }
  }
}

// ---------------------------------------------------------------------------
extern "C" void kernel_launch(void* const* d_in, const int* in_sizes, int n_in,
                              void* d_out, int out_size, void* d_ws, size_t ws_size,
                              hipStream_t stream) {
  const float* x   = (const float*)d_in[0];
  const float* ctx = (const float*)d_in[1];
  const float* Wq  = (const float*)d_in[2];
  const float* bq  = (const float*)d_in[3];
  const float* Wk  = (const float*)d_in[4];
  const float* bk  = (const float*)d_in[5];
  const float* Wv  = (const float*)d_in[6];
  const float* bv  = (const float*)d_in[7];
  const float* Wo  = (const float*)d_in[8];
  const float* bo  = (const float*)d_in[9];
  float* out = (float*)d_out;

  // Workspace layout (bytes):
  //   Wt : 4*512*512*2        =  2,097,152  @ 0
  //   Qs : 4*16*2048*32*2     =  8,388,608  @ 2,097,152
  //   Ks : same               =  8,388,608  @ 10,485,760
  //   Vt : same (transposed)  =  8,388,608  @ 18,874,368
  //   AO : 8192*512*2         =  8,388,608  @ 27,262,976   (total ~34 MB)
  char* ws = (char*)d_ws;
  __bf16* Wt = (__bf16*)ws;
  __bf16* Qs = (__bf16*)(ws + 2097152);
  __bf16* Ks = (__bf16*)(ws + 10485760);
  __bf16* Vt = (__bf16*)(ws + 18874368);
  __bf16* AO = (__bf16*)(ws + 27262976);

  k_prep<<<dim3(16, 16, 4), dim3(32, 8, 1), 0, stream>>>(Wq, Wk, Wv, Wo, Wt);
  k_qkv<<<3072, 128, 0, stream>>>(x, ctx, Wt, bq, bk, bv, Qs, Ks, Vt);
  k_attn<<<2048, 128, 0, stream>>>(Qs, Ks, Vt, AO);
  k_oproj<<<1024, 128, 0, stream>>>(AO, Wt, bo, out);
}
